// StructureModuleLayer_11501922419431
// MI455X (gfx1250) — matrix operand order repaired and checked by reading, not compiled, not run on hardware
//
#include <hip/hip_runtime.h>
#include <math.h>

// ---------------------------------------------------------------------------
// StructureModule layer for MI455X (gfx1250, wave32, WMMA).
// All GEMM-shaped work goes through v_wmma_f32_16x16x32_f16 (f32 accumulate).
// Weights are pre-packed into f16 B-fragment layout (contiguous 32B per lane
// per k-step); attention/bias buffers are head-major for contiguous fragments.
// ---------------------------------------------------------------------------

typedef __attribute__((ext_vector_type(16))) _Float16 v16h;
typedef __attribute__((ext_vector_type(8)))  float    v8f;

#define NN   512
#define CSD  384
#define CZD  128
#define HH   12
#define PVD  8
#define PROJW 1152            // 192*3 + 144*2 + 288
#define CATW  2112            // 192 + 288 + 96 + 1536
#define NSQ  (NN * NN)        // 262144

__device__ __constant__ float WLc = 0.57735026918962576f;  // sqrt(1/3)
__device__ __constant__ float WCc = 0.23570226039551584f;  // sqrt(2/36)

// --- WMMA fragment index helpers (CDNA5 ISA 7.12.2, 16-bit A 16x32) --------
// A: lane L holds row (L&15); element e -> K = ((L>>4)<<4) + e (this code's order; ISA table differs)
__device__ __forceinline__ int a_kidx(int lane, int e) {
  return ((lane >> 4) << 4) + e;
}
// B: lane L holds col (L&15); element e -> K = ((L>>4)<<4) + e (this code's order, matching A)
// C/D: lane L col (L&15); VGPR v -> row ((L>>4)<<3) + v

__device__ __forceinline__ v16h load_a_frag(const float* __restrict__ A,
                                            int lda, int row0, int k0) {
  int lane = threadIdx.x & 31;
  const float* p = A + (size_t)(row0 + (lane & 15)) * lda + k0;
  v16h a;
#pragma unroll
  for (int e = 0; e < 16; ++e) a[e] = (_Float16)p[a_kidx(lane, e)];
  return a;
}

__device__ __forceinline__ float wave_sum(float v) {
#pragma unroll
  for (int off = 16; off > 0; off >>= 1) v += __shfl_xor(v, off, 32);
  return v;
}
__device__ __forceinline__ float wave_max(float v) {
#pragma unroll
  for (int off = 16; off > 0; off >>= 1) v = fmaxf(v, __shfl_xor(v, off, 32));
  return v;
}

// ---------------------------------------------------------------------------
// Pack a weight matrix W (K x N, row-major, K%32==0) into B-fragment layout:
// halves laid out as [(nt*Kt + kt)*32 + lane][16], zero-padded past N.
// ---------------------------------------------------------------------------
__global__ void packB_kernel(const float* __restrict__ W, _Float16* __restrict__ Bp,
                             int N, int K, int ldb) {
  int Kt = K >> 5;
  int ntiles = (N + 15) >> 4;
  int total = ntiles * Kt * 32;
  int tid = blockIdx.x * blockDim.x + threadIdx.x;
  if (tid >= total) return;
  int lane = tid & 31;
  int kt = (tid >> 5) % Kt;
  int nt = (tid >> 5) / Kt;
  int n  = nt * 16 + (lane & 15);
  int kb = kt * 32 + ((lane >> 4) << 4);
  _Float16* dst = Bp + (size_t)tid * 16;
#pragma unroll
  for (int e = 0; e < 16; ++e)
    dst[e] = (n < N) ? (_Float16)W[(size_t)(kb + e) * ldb + n] : (_Float16)0.0f;
}

// Pack per-head [v (16 cols) | gv (24 cols, padded to 32)] into fragments:
// layout [((h*3 + nt)*16 + kt)*32 + lane][16], Kt = 16 (K = 512).
__global__ void pack_vgv_kernel(const float* __restrict__ proj,
                                const float* __restrict__ gv,
                                _Float16* __restrict__ Bp) {
  int tid = blockIdx.x * blockDim.x + threadIdx.x;
  if (tid >= HH * 3 * 16 * 32) return;
  int lane = tid & 31;
  int kt = (tid >> 5) % 16;
  int r  = (tid >> 5) / 16;
  int nt = r % 3, h = r / 3;
  int kb = kt * 32 + ((lane >> 4) << 4);
  int nc = lane & 15;
  _Float16* dst = Bp + (size_t)tid * 16;
#pragma unroll
  for (int e = 0; e < 16; ++e) {
    int j = kb + e;
    float v;
    if (nt == 0) v = proj[(size_t)j * PROJW + 384 + h * 16 + nc];
    else {
      int col = (nt - 1) * 16 + nc;
      v = (col < 24) ? gv[(size_t)(j * HH + h) * 24 + col] : 0.0f;
    }
    dst[e] = (_Float16)v;
  }
}

// ---------------------------------------------------------------------------
// Packed-B WMMA GEMM: C[r*rs + n*cs] = op(A@Bp + bias + res).
// flags: 1=relu, 2=add bias[n], 4=add residual.  M%16==0, K%32==0.
// One wave per 16x16 tile; B loads are one contiguous 32B v16h per lane.
// ---------------------------------------------------------------------------
__global__ void gemm_wmma_pk(const float* __restrict__ A, const _Float16* __restrict__ Bp,
                             const float* __restrict__ bias, const float* __restrict__ res,
                             float* __restrict__ C,
                             int M, int N, int K, int lda, int rs, int cs, int flags) {
  int wave   = blockIdx.x * (blockDim.x >> 5) + (threadIdx.x >> 5);
  int ntiles = (N + 15) >> 4;
  int mt = wave / ntiles, nt = wave % ntiles;
  if (mt * 16 >= M) return;
  int lane = threadIdx.x & 31;
  int row0 = mt * 16;
  int Kt = K >> 5;
  const _Float16* bptr = Bp + ((size_t)nt * Kt * 32 + lane) * 16;
  const float* aptr = A + (size_t)(row0 + (lane & 15)) * lda;

  v8f acc = {};
  for (int kt = 0; kt < Kt; ++kt) {
    v16h a;
    const float* p = aptr + kt * 32;
#pragma unroll
    for (int e = 0; e < 16; ++e) a[e] = (_Float16)p[a_kidx(lane, e)];
    if (kt + 1 < Kt) __builtin_prefetch(p + 32, 0, 1);
    v16h b = *(const v16h*)(bptr + (size_t)kt * 32 * 16);
    acc = __builtin_amdgcn_wmma_f32_16x16x32_f16(false, a, false, b,
                                                 (short)0, acc, false, false);
  }
  int n = nt * 16 + (lane & 15);
  if (n < N) {
    float bv = (flags & 2) ? bias[n] : 0.0f;
    int rb = row0 + ((lane >> 4) << 3);
#pragma unroll
    for (int v = 0; v < 8; ++v) {
      size_t idx = (size_t)(rb + v) * rs + (size_t)n * cs;
      float x = acc[v] + bv;
      if (flags & 4) x += res[idx];
      if (flags & 1) x = fmaxf(x, 0.0f);
      C[idx] = x;
    }
  }
}

// ---------------------------------------------------------------------------
// Frames: rotate/translate qp,kp,vp -> gq,gk,gv.  Thread per (i,h,point-slot).
// ---------------------------------------------------------------------------
__global__ void frames_kernel(const float* __restrict__ proj,
                              const float* __restrict__ rot,
                              const float* __restrict__ trans,
                              float* __restrict__ gq, float* __restrict__ gk,
                              float* __restrict__ gv) {
  int tid = blockIdx.x * blockDim.x + threadIdx.x;
  if (tid >= NN * HH * 16) return;
  int i = tid / (HH * 16);
  int r = tid % (HH * 16);
  int h = r / 16, s = r % 16;
  const float* R = rot + (size_t)i * 9;
  const float* T = trans + (size_t)i * 3;
  int base; float* dst;
  if (s < 4)      { int p = s;     base = 576 + h * 12 + p * 3; dst = gq + ((size_t)(i*HH+h)*12 + p*3); }
  else if (s < 8) { int p = s - 4; base = 720 + h * 12 + p * 3; dst = gk + ((size_t)(i*HH+h)*12 + p*3); }
  else            { int p = s - 8; base = 864 + h * 24 + p * 3; dst = gv + ((size_t)(i*HH+h)*24 + p*3); }
  const float* src = proj + (size_t)i * PROJW + base;
  float x0 = src[0], x1 = src[1], x2 = src[2];
#pragma unroll
  for (int a = 0; a < 3; ++a)
    dst[a] = R[a*3+0]*x0 + R[a*3+1]*x1 + R[a*3+2]*x2 + T[a];
}

__global__ void sq_kernel(const float* __restrict__ gq, const float* __restrict__ gk,
                          float* __restrict__ sqq, float* __restrict__ sqk) {
  int tid = blockIdx.x * blockDim.x + threadIdx.x;
  if (tid >= NN * HH) return;
  float s = 0.0f;
#pragma unroll
  for (int e = 0; e < 12; ++e) { float x = gq[(size_t)tid * 12 + e]; s += x * x; }
  sqq[tid] = s;
  s = 0.0f;
#pragma unroll
  for (int e = 0; e < 12; ++e) { float x = gk[(size_t)tid * 12 + e]; s += x * x; }
  sqk[tid] = s;
}

// ---------------------------------------------------------------------------
// Logits: single K=28(->32) WMMA packs qk and point-cross; epilogue adds
// bias and rank-1 terms.  head-major output att[h][i][j].
// ---------------------------------------------------------------------------
__global__ void logits_kernel(const float* __restrict__ proj,
                              const float* __restrict__ gq, const float* __restrict__ gk,
                              const float* __restrict__ sqq, const float* __restrict__ sqk,
                              const float* __restrict__ bias,   // [h][i*NN+j]
                              const float* __restrict__ gamma_raw,
                              float* __restrict__ att) {        // [h][i*NN+j]
  int wave = blockIdx.x * (blockDim.x >> 5) + (threadIdx.x >> 5);
  if (wave >= HH * 32 * 32) return;
  int h   = wave / 1024;
  int rem = wave % 1024;
  int it = rem >> 5, jt = rem & 31;
  int lane = threadIdx.x & 31;

  float gr    = gamma_raw[h];
  float gamma = (gr > 20.0f) ? gr : log1pf(expf(gr));
  float cq = WLc * 0.25f;          // WL * C^-0.5
  float cg = WLc * WCc * gamma;    // coefficient on cross term

  int ri = it * 16 + (lane & 15);
  int cj = jt * 16 + (lane & 15);

  v16h a, b;
#pragma unroll
  for (int e = 0; e < 16; ++e) {
    int kk = a_kidx(lane, e);
    float av;
    if (kk < 16)      av = proj[(size_t)ri * PROJW + h * 16 + kk] * cq;
    else if (kk < 28) av = gq[(size_t)(ri * HH + h) * 12 + (kk - 16)] * cg;
    else              av = 0.0f;
    a[e] = (_Float16)av;
  }
  int kb = (lane >> 4) << 4;
#pragma unroll
  for (int e = 0; e < 16; ++e) {
    int kk = kb + e;
    float bv;
    if (kk < 16)      bv = proj[(size_t)cj * PROJW + 192 + h * 16 + kk];
    else if (kk < 28) bv = gk[(size_t)(cj * HH + h) * 12 + (kk - 16)];
    else              bv = 0.0f;
    b[e] = (_Float16)bv;
  }
  v8f acc = {};
  acc = __builtin_amdgcn_wmma_f32_16x16x32_f16(false, a, false, b,
                                               (short)0, acc, false, false);
  float skj = sqk[cj * HH + h];
  float c2  = 0.5f * WLc * WCc * gamma;
  int rb = it * 16 + ((lane >> 4) << 3);
  const float* bh = bias + (size_t)h * NSQ;
  float* ah = att + (size_t)h * NSQ;
#pragma unroll
  for (int v = 0; v < 8; ++v) {
    int i = rb + v;
    float l = acc[v] + WLc * bh[(size_t)i * NN + cj]
                     - c2 * (sqq[i * HH + h] + skj);
    ah[(size_t)i * NN + cj] = l;
  }
}

// Softmax over j for each (h,i): one wave reads 512 contiguous floats.
__global__ void softmax_kernel(float* __restrict__ att) {
  int wid = blockIdx.x * (blockDim.x >> 5) + (threadIdx.x >> 5);
  if (wid >= NN * HH) return;
  float* row = att + (size_t)wid * NN;
  int lane = threadIdx.x & 31;
  float vals[16];
  float m = -1e30f;
#pragma unroll
  for (int t = 0; t < 16; ++t) {
    vals[t] = row[lane + t * 32];
    m = fmaxf(m, vals[t]);
  }
  m = wave_max(m);
  float s = 0.0f;
#pragma unroll
  for (int t = 0; t < 16; ++t) { vals[t] = expf(vals[t] - m); s += vals[t]; }
  s = wave_sum(s);
  float inv = 1.0f / s;
#pragma unroll
  for (int t = 0; t < 16; ++t) row[lane + t * 32] = vals[t] * inv;
}

// ---------------------------------------------------------------------------
// o_v and o_pt: per-head GEMM  A_h(512x512) @ packed [v_h | gv_h] fragments.
// One wave per (h, itile, ntile{0,1,2}).
// ---------------------------------------------------------------------------
__global__ void ovpt_kernel(const float* __restrict__ att, const _Float16* __restrict__ Bp,
                            float* __restrict__ cat, float* __restrict__ opt) {
  int wave = blockIdx.x * (blockDim.x >> 5) + (threadIdx.x >> 5);
  if (wave >= HH * 32 * 3) return;
  int h = wave / 96, rem = wave % 96;
  int it = rem / 3, nt = rem % 3;
  int lane = threadIdx.x & 31;
  int ncol = lane & 15;
  const float* arow = att + (size_t)h * NSQ + (size_t)(it * 16 + (lane & 15)) * NN;
  const _Float16* bptr = Bp + ((size_t)((h * 3 + nt) * 16) * 32 + lane) * 16;

  v8f acc = {};
  for (int kt = 0; kt < 16; ++kt) {
    v16h a;
    const float* p = arow + kt * 32;
#pragma unroll
    for (int e = 0; e < 16; ++e) a[e] = (_Float16)p[a_kidx(lane, e)];
    v16h b = *(const v16h*)(bptr + (size_t)kt * 32 * 16);
    acc = __builtin_amdgcn_wmma_f32_16x16x32_f16(false, a, false, b,
                                                 (short)0, acc, false, false);
  }
  int rb = it * 16 + ((lane >> 4) << 3);
#pragma unroll
  for (int v = 0; v < 8; ++v) {
    int i = rb + v;
    if (nt == 0) {
      cat[(size_t)i * CATW + h * 16 + ncol] = acc[v];
    } else {
      int col = (nt - 1) * 16 + ncol;
      if (col < 24) opt[(size_t)(i * HH + h) * 24 + col] = acc[v];
    }
  }
}

// o_pair: per-row batched GEMM  a[i]^T(12x512) @ pair[i](512x128).
// One wave per (i, ntile of 8); pair stays L2-resident after bias pass.
__global__ void opair_kernel(const float* __restrict__ att, const float* __restrict__ pair,
                             float* __restrict__ cat) {
  int wave = blockIdx.x * (blockDim.x >> 5) + (threadIdx.x >> 5);
  if (wave >= NN * 8) return;
  int i = wave >> 3, nt = wave & 7;
  int lane = threadIdx.x & 31;
  int hrow = (lane & 15) < HH ? (lane & 15) : (HH - 1);   // clamp padded head rows
  int ncol = nt * 16 + (lane & 15);
  const float* prow = pair + (size_t)i * NN * CZD;
  const float* arow = att + (size_t)hrow * NSQ + (size_t)i * NN;

  v8f acc = {};
  for (int k0 = 0; k0 < NN; k0 += 32) {
    v16h a, b;
    const float* p = arow + k0;
#pragma unroll
    for (int e = 0; e < 16; ++e) a[e] = (_Float16)p[a_kidx(lane, e)];
    int kb = k0 + ((lane >> 4) << 4);
#pragma unroll
    for (int e = 0; e < 16; ++e)
      b[e] = (_Float16)prow[(size_t)(kb + e) * CZD + ncol];
    if (k0 + 32 < NN)
      __builtin_prefetch(prow + (size_t)(kb + 32) * CZD + ncol, 0, 1);
    acc = __builtin_amdgcn_wmma_f32_16x16x32_f16(false, a, false, b,
                                                 (short)0, acc, false, false);
  }
  int rb = (lane >> 4) << 3;
#pragma unroll
  for (int v = 0; v < 8; ++v) {
    int hh = rb + v;
    if (hh < HH) cat[(size_t)i * CATW + 576 + hh * CZD + ncol] = acc[v];
  }
}

// local = rot^T (o_pt - trans); norms.  Thread per (i,h,p).
__global__ void localnorm_kernel(const float* __restrict__ opt,
                                 const float* __restrict__ rot,
                                 const float* __restrict__ trans,
                                 float* __restrict__ cat) {
  int tid = blockIdx.x * blockDim.x + threadIdx.x;
  if (tid >= NN * HH * PVD) return;
  int i = tid / (HH * PVD);
  int r = tid % (HH * PVD);
  int h = r / PVD, p = r % PVD;
  const float* R = rot + (size_t)i * 9;
  const float* T = trans + (size_t)i * 3;
  float d0 = opt[(size_t)tid * 3 + 0] - T[0];
  float d1 = opt[(size_t)tid * 3 + 1] - T[1];
  float d2 = opt[(size_t)tid * 3 + 2] - T[2];
  float l[3], nsq = 1e-8f;
#pragma unroll
  for (int a = 0; a < 3; ++a) {
    l[a] = R[0*3+a]*d0 + R[1*3+a]*d1 + R[2*3+a]*d2;
    nsq += l[a] * l[a];
    cat[(size_t)i * CATW + 192 + h * 24 + p * 3 + a] = l[a];
  }
  cat[(size_t)i * CATW + 480 + h * 8 + p] = sqrtf(nsq);
}

// LayerNorm: one wave per row of 384.
__global__ void layernorm_kernel(const float* __restrict__ x, const float* __restrict__ g,
                                 const float* __restrict__ b, float* __restrict__ y) {
  int wid = blockIdx.x * (blockDim.x >> 5) + (threadIdx.x >> 5);
  if (wid >= NN) return;
  int lane = threadIdx.x & 31;
  float vals[12];
  float s = 0.0f;
#pragma unroll
  for (int t = 0; t < 12; ++t) { vals[t] = x[(size_t)wid * CSD + lane + t * 32]; s += vals[t]; }
  float mean = wave_sum(s) * (1.0f / CSD);
  float v = 0.0f;
#pragma unroll
  for (int t = 0; t < 12; ++t) { float d = vals[t] - mean; v += d * d; }
  float inv = rsqrtf(wave_sum(v) * (1.0f / CSD) + 1e-5f);
#pragma unroll
  for (int t = 0; t < 12; ++t) {
    int c = lane + t * 32;
    y[(size_t)wid * CSD + c] = (vals[t] - mean) * inv * g[c] + b[c];
  }
}

// Backbone update: one wave per residue; also zeros the loss slot.
__global__ void backbone_kernel(const float* __restrict__ sfin,
                                const float* __restrict__ Wbu, const float* __restrict__ bbu,
                                const float* __restrict__ rot, const float* __restrict__ trans,
                                float* __restrict__ out_rot, float* __restrict__ out_trans,
                                float* __restrict__ out_loss) {
  int wid = blockIdx.x * (blockDim.x >> 5) + (threadIdx.x >> 5);
  if (wid >= NN) return;
  int lane = threadIdx.x & 31;
  float acc[6] = {0, 0, 0, 0, 0, 0};
#pragma unroll
  for (int t = 0; t < 12; ++t) {
    int c = lane + t * 32;
    float s = sfin[(size_t)wid * CSD + c];
#pragma unroll
    for (int o = 0; o < 6; ++o) acc[o] += s * Wbu[c * 6 + o];
  }
#pragma unroll
  for (int o = 0; o < 6; ++o) acc[o] = wave_sum(acc[o]);
  if (lane == 0) {
    float u[6];
#pragma unroll
    for (int o = 0; o < 6; ++o) u[o] = acc[o] + bbu[o];
    float qa = 1.0f, qb = u[0], qc = u[1], qd = u[2];
    float inv = rsqrtf(qa*qa + qb*qb + qc*qc + qd*qd);
    qa *= inv; qb *= inv; qc *= inv; qd *= inv;
    float Rm[9];
    Rm[0] = qa*qa + qb*qb - qc*qc - qd*qd; Rm[1] = 2.0f*(qb*qc - qa*qd); Rm[2] = 2.0f*(qb*qd + qa*qc);
    Rm[3] = 2.0f*(qb*qc + qa*qd); Rm[4] = qa*qa - qb*qb + qc*qc - qd*qd; Rm[5] = 2.0f*(qc*qd - qa*qb);
    Rm[6] = 2.0f*(qb*qd - qa*qc); Rm[7] = 2.0f*(qc*qd + qa*qb); Rm[8] = qa*qa - qb*qb - qc*qc + qd*qd;
    const float* Ro = rot + (size_t)wid * 9;
#pragma unroll
    for (int a = 0; a < 3; ++a)
#pragma unroll
      for (int c = 0; c < 3; ++c)
        out_rot[(size_t)wid * 9 + a * 3 + c] =
            Ro[a*3+0]*Rm[0*3+c] + Ro[a*3+1]*Rm[1*3+c] + Ro[a*3+2]*Rm[2*3+c];
    const float* T = trans + (size_t)wid * 3;
#pragma unroll
    for (int a = 0; a < 3; ++a)
      out_trans[(size_t)wid * 3 + a] =
          Ro[a*3+0]*u[3] + Ro[a*3+1]*u[4] + Ro[a*3+2]*u[5] + T[a];
    if (wid == 0) *out_loss = 0.0f;
  }
}

// Clamped FAPE loss: thread per (i,j) pair, block-reduced atomicAdd.
__global__ void loss_kernel(const float* __restrict__ rot_t, const float* __restrict__ trans_t,
                            const float* __restrict__ nrot, const float* __restrict__ ntrans,
                            float* __restrict__ out_loss) {
  int p = blockIdx.x * blockDim.x + threadIdx.x;
  int i = p >> 9, j = p & 511;
  const float* Ri  = nrot + (size_t)i * 9;
  const float* Rti = rot_t + (size_t)i * 9;
  float d0 = ntrans[j*3+0] - ntrans[i*3+0];
  float d1 = ntrans[j*3+1] - ntrans[i*3+1];
  float d2 = ntrans[j*3+2] - ntrans[i*3+2];
  float e0 = trans_t[j*3+0] - trans_t[i*3+0];
  float e1 = trans_t[j*3+1] - trans_t[i*3+1];
  float e2 = trans_t[j*3+2] - trans_t[i*3+2];
  float acc = 1e-12f;
#pragma unroll
  for (int a = 0; a < 3; ++a) {
    float dp = Ri[0*3+a]*d0 + Ri[1*3+a]*d1 + Ri[2*3+a]*d2;   // 'iba,ijb->ija'
    float dt = Rti[0*3+a]*e0 + Rti[1*3+a]*e1 + Rti[2*3+a]*e2;
    float df = dp - dt;
    acc += df * df;
  }
  float v = fminf(sqrtf(acc), 10.0f) * (1.0f / (10.0f * (float)NSQ));
  v = wave_sum(v);
  __shared__ float red[8];
  int w = threadIdx.x >> 5;
  if ((threadIdx.x & 31) == 0) red[w] = v;
  __syncthreads();
  if (threadIdx.x == 0) {
    float s = 0.0f;
#pragma unroll
    for (int q = 0; q < 8; ++q) s += red[q];
    atomicAdd(out_loss, s);
  }
}

// ---------------------------------------------------------------------------
extern "C" void kernel_launch(void* const* d_in, const int* in_sizes, int n_in,
                              void* d_out, int out_size, void* d_ws, size_t ws_size,
                              hipStream_t stream) {
  const float* single_in  = (const float*)d_in[0];
  const float* pair       = (const float*)d_in[1];
  const float* rot        = (const float*)d_in[2];
  const float* trans      = (const float*)d_in[3];
  const float* rot_truth  = (const float*)d_in[4];
  const float* trans_trth = (const float*)d_in[5];
  const float* Wq  = (const float*)d_in[6];
  const float* Wk  = (const float*)d_in[7];
  const float* Wv  = (const float*)d_in[8];
  const float* Wqp = (const float*)d_in[9];
  const float* Wkp = (const float*)d_in[10];
  const float* Wvp = (const float*)d_in[11];
  const float* Wb  = (const float*)d_in[12];
  const float* Wo  = (const float*)d_in[13];
  const float* bo  = (const float*)d_in[14];
  const float* gamma_raw = (const float*)d_in[15];
  const float* ln1_g = (const float*)d_in[16];
  const float* ln1_b = (const float*)d_in[17];
  const float* ln2_g = (const float*)d_in[18];
  const float* ln2_b = (const float*)d_in[19];
  const float* W1 = (const float*)d_in[20];
  const float* b1 = (const float*)d_in[21];
  const float* W2 = (const float*)d_in[22];
  const float* b2 = (const float*)d_in[23];
  const float* W3 = (const float*)d_in[24];
  const float* b3 = (const float*)d_in[25];
  const float* Wbu = (const float*)d_in[26];
  const float* bbu = (const float*)d_in[27];

  // ---- Workspace layout: float region (all sizes multiples of 16) ---------
  float* ws    = (float*)d_ws;
  float* proj  = ws;                                  // 512*1152
  float* gqB   = proj + (size_t)NN * PROJW;           // 512*12*12
  float* gkB   = gqB + (size_t)NN * HH * 12;
  float* gvB   = gkB + (size_t)NN * HH * 12;          // 512*12*24
  float* sqqB  = gvB + (size_t)NN * HH * 24;          // 512*12
  float* sqkB  = sqqB + (size_t)NN * HH;
  float* biasB = sqkB + (size_t)NN * HH;              // [h][ij] 12*262144
  float* attB  = biasB + (size_t)HH * NSQ;            // [h][ij] 12*262144
  float* optB  = attB + (size_t)HH * NSQ;             // 512*12*24
  float* catB  = optB + (size_t)NN * HH * 24;         // 512*2112
  float* s1    = catB + (size_t)NN * CATW;            // 512*384
  float* slnB  = s1 + (size_t)NN * CSD;
  float* h1    = slnB + (size_t)NN * CSD;
  float* h2    = h1 + (size_t)NN * CSD;
  float* s2    = h2 + (size_t)NN * CSD;

  // ---- Packed f16 B-fragment region --------------------------------------
  _Float16* hp    = (_Float16*)(s2 + (size_t)NN * CSD);
  _Float16* pkWq  = hp;                   // 192*384
  _Float16* pkWk  = pkWq + 73728;
  _Float16* pkWv  = pkWk + 73728;
  _Float16* pkWqp = pkWv + 73728;         // 144*384
  _Float16* pkWkp = pkWqp + 55296;
  _Float16* pkWvp = pkWkp + 55296;        // 288*384
  _Float16* pkWb  = pkWvp + 110592;       // 16*128
  _Float16* pkWo  = pkWb + 2048;          // 384*2112
  _Float16* pkW1  = pkWo + 811008;        // 384*384
  _Float16* pkW2  = pkW1 + 147456;
  _Float16* pkW3  = pkW2 + 147456;
  _Float16* pkVgv = pkW3 + 147456;        // 12*3*16*32*16

  float* out        = (float*)d_out;
  float* out_single = out;                            // 512*384
  float* out_rot    = out + (size_t)NN * CSD;         // 512*9
  float* out_trans  = out_rot + (size_t)NN * 9;       // 512*3
  float* out_loss   = out_trans + (size_t)NN * 3;     // 1

  auto pack = [&](const float* W, _Float16* Bp, int N, int K, int ldb) {
    int total = ((N + 15) / 16) * (K / 32) * 32;
    packB_kernel<<<(total + 255) / 256, 256, 0, stream>>>(W, Bp, N, K, ldb);
  };
  auto gemm = [&](const float* A, const _Float16* Bp, const float* bias, const float* res,
                  float* C, int M, int N, int K, int lda, int rs, int cs, int flags) {
    int tiles = ((M + 15) / 16) * ((N + 15) / 16);
    gemm_wmma_pk<<<(tiles + 3) / 4, 128, 0, stream>>>(A, Bp, bias, res, C,
                                                      M, N, K, lda, rs, cs, flags);
  };

  // 0) Pack all weight matrices into B-fragment layout (~4 MB, L2-resident)
  pack(Wq,  pkWq,  192, CSD, 192);
  pack(Wk,  pkWk,  192, CSD, 192);
  pack(Wv,  pkWv,  192, CSD, 192);
  pack(Wqp, pkWqp, 144, CSD, 144);
  pack(Wkp, pkWkp, 144, CSD, 144);
  pack(Wvp, pkWvp, 288, CSD, 288);
  pack(Wb,  pkWb,  HH,  CZD, HH);
  pack(Wo,  pkWo,  CSD, CATW, CSD);
  pack(W1,  pkW1,  CSD, CSD, CSD);
  pack(W2,  pkW2,  CSD, CSD, CSD);
  pack(W3,  pkW3,  CSD, CSD, CSD);

  // 1) Fused projections -> proj (512 x 1152)
  gemm(single_in, pkWq,  nullptr, nullptr, proj + 0,   NN, 192, CSD, CSD, PROJW, 1, 0);
  gemm(single_in, pkWk,  nullptr, nullptr, proj + 192, NN, 192, CSD, CSD, PROJW, 1, 0);
  gemm(single_in, pkWv,  nullptr, nullptr, proj + 384, NN, 192, CSD, CSD, PROJW, 1, 0);
  gemm(single_in, pkWqp, nullptr, nullptr, proj + 576, NN, 144, CSD, CSD, PROJW, 1, 0);
  gemm(single_in, pkWkp, nullptr, nullptr, proj + 720, NN, 144, CSD, CSD, PROJW, 1, 0);
  gemm(single_in, pkWvp, nullptr, nullptr, proj + 864, NN, 288, CSD, CSD, PROJW, 1, 0);

  // 2) Global frames + squared norms; pack [v|gv] fragments per head
  frames_kernel<<<(NN * HH * 16 + 255) / 256, 256, 0, stream>>>(proj, rot, trans, gqB, gkB, gvB);
  sq_kernel<<<(NN * HH + 255) / 256, 256, 0, stream>>>(gqB, gkB, sqqB, sqkB);
  pack_vgv_kernel<<<(HH * 3 * 16 * 32 + 255) / 256, 256, 0, stream>>>(proj, gvB, pkVgv);

  // 3) Pair bias: (512*512 x 128) @ (128 x 12), head-major store.
  //    This is the single HBM pass over the 128 MB pair tensor -> L2.
  gemm(pair, pkWb, nullptr, nullptr, biasB, NSQ, HH, CZD, CZD, 1, NSQ, 0);

  // 4) Attention logits (fused qk + point-cross WMMA) and softmax
  logits_kernel<<<(HH * 32 * 32) / 4, 128, 0, stream>>>(proj, gqB, gkB, sqqB, sqkB,
                                                        biasB, gamma_raw, attB);
  softmax_kernel<<<(NN * HH) / 4, 128, 0, stream>>>(attB);

  // 5) Attention outputs
  ovpt_kernel<<<(HH * 32 * 3 + 3) / 4, 128, 0, stream>>>(attB, pkVgv, catB, optB);
  opair_kernel<<<(NN * 8) / 4, 128, 0, stream>>>(attB, pair, catB);   // pair hits L2
  localnorm_kernel<<<(NN * HH * PVD + 255) / 256, 256, 0, stream>>>(optB, rot, trans, catB);

  // 6) Output projection + residual, LN1
  gemm(catB, pkWo, bo, single_in, s1, NN, CSD, CATW, CATW, CSD, 1, 2 | 4);
  layernorm_kernel<<<NN / 4, 128, 0, stream>>>(s1, ln1_g, ln1_b, slnB);

  // 7) Transition MLP + residual, LN2 -> output single
  gemm(slnB, pkW1, b1, nullptr, h1, NN, CSD, CSD, CSD, CSD, 1, 1 | 2);
  gemm(h1,   pkW2, b2, nullptr, h2, NN, CSD, CSD, CSD, CSD, 1, 1 | 2);
  gemm(h2,   pkW3, b3, slnB,    s2, NN, CSD, CSD, CSD, CSD, 1, 2 | 4);
  layernorm_kernel<<<NN / 4, 128, 0, stream>>>(s2, ln2_g, ln2_b, out_single);

  // 8) Backbone update (quats) and clamped FAPE loss
  backbone_kernel<<<NN / 4, 128, 0, stream>>>(out_single, Wbu, bbu, rot, trans,
                                              out_rot, out_trans, out_loss);
  loss_kernel<<<NSQ / 256, 256, 0, stream>>>(rot_truth, trans_trth,
                                             out_rot, out_trans, out_loss);
}
